// ModConv2d_86792699118234
// MI455X (gfx1250) — matrix operand order repaired and checked by reading, not compiled, not run on hardware
//
#include <hip/hip_runtime.h>

typedef __attribute__((ext_vector_type(16))) _Float16 v16h;
typedef __attribute__((ext_vector_type(8)))  _Float16 v8h;
typedef __attribute__((ext_vector_type(4)))  _Float16 v4h;
typedef __attribute__((ext_vector_type(8)))  float    v8f;

#define KTOT   1152                          // 9 * 128
#define NKITER 36                            // KTOT / 32
#define ROWSTRIDE 136                        // 128 + 8 halves pad -> conflict-free A reads
#define PCOLS  258                           // 256 pixels + 2 halo columns
#define PATCH_HALVES (3 * PCOLS * ROWSTRIDE) // 105264 halves = 210528 B
#define WTAP_HALVES  (4 * 4096)              // one 3x3-tap of weights: 16384 halves = 32 KB

#if defined(__has_builtin)
#  if __has_builtin(__builtin_amdgcn_global_load_async_to_lds_b128)
#    define ASYNC_WSTAGE 1
#  endif
#endif

#if defined(ASYNC_WSTAGE)
// Builtin signature (from hipcc diagnostic): param0 = non-const v4i __device__(AS1)*,
// param1 = v4i local(AS3)*, then imm offset, imm cpol.
typedef int vi4 __attribute__((vector_size(16)));
typedef __attribute__((address_space(1))) vi4 gvi4;
typedef __attribute__((address_space(3))) vi4 svi4;

__device__ __forceinline__ void wstage_async(const _Float16* src, _Float16* dst, int tid) {
    #pragma unroll
    for (int j = 0; j < 8; ++j) {
        gvi4* gs = (gvi4*)((const vi4*)src + tid + j * 256);  // strip const + AS cast
        svi4* ls = (svi4*)((vi4*)dst + tid + j * 256);
        __builtin_amdgcn_global_load_async_to_lds_b128(gs, ls, 0, 0);
    }
}
__device__ __forceinline__ void wstage_wait() {
#  if __has_builtin(__builtin_amdgcn_s_wait_asynccnt)
    __builtin_amdgcn_s_wait_asynccnt(0);
#  else
    asm volatile("s_wait_asynccnt 0x0" ::: "memory");
#  endif
}
#endif

// ---------------- Prep: build modulated f16 weights in WMMA-B fragment order ----------------
// wfrag layout: chunk c = ((kiter*8 + ntile)*32 + lane), 16 halves per chunk.
// Within a chunk: n = ntile*16 + (lane&15), k = kiter*32 + (lane>>4)*16 + i.
// Tap kk occupies the contiguous range [kk*WTAP_HALVES, (kk+1)*WTAP_HALVES).
__global__ __launch_bounds__(256) void modconv_prep(
    const float* __restrict__ style, const float* __restrict__ kern,
    const float* __restrict__ w_mod, const float* __restrict__ b_mod,
    _Float16* __restrict__ wfrag)
{
    __shared__ float s_lds[1024];
    __shared__ float s0_lds[128];
    __shared__ float d_lds[128];
    const int tid = threadIdx.x;

    for (int j = 0; j < 4; ++j) {
        int e = tid + j * 256;
        int b = e >> 7, cin = e & 127;
        float acc = 0.f;
        for (int q = 0; q < 512; ++q)
            acc = fmaf(style[b * 512 + q], w_mod[q * 128 + cin], acc);
        s_lds[e] = acc + b_mod[cin] + 1.0f;
    }
    __syncthreads();

    float m = 0.f;
    for (int i = 0; i < 1024; ++i) m = fmaxf(m, fabsf(s_lds[i]));
    float inv = 1.0f / m;
    if (tid < 128) s0_lds[tid] = s_lds[tid] * inv;   // s[0] row only
    __syncthreads();

    const float he_std = 0.02946278254943948f;       // 1/sqrt(9*128)
    if (tid < 128) {
        float sum = 0.f;
        for (int i = 0; i < KTOT; ++i) {
            float v = kern[i * 128 + tid] * he_std * s0_lds[i & 127];
            sum = fmaf(v, v, sum);
        }
        d_lds[tid] = rsqrtf(sum + 1e-8f);
    }
    __syncthreads();

    for (int c = tid; c < NKITER * 8 * 32; c += 256) {
        int kiter = c >> 8;
        int rem   = c & 255;
        int nt    = rem >> 5;
        int lane  = rem & 31;
        int n     = nt * 16 + (lane & 15);
        int kb    = kiter * 32 + (lane >> 4) * 16;
        float dn  = d_lds[n];
        _Float16* outp = wfrag + (size_t)c * 16;
        for (int i = 0; i < 16; ++i) {
            int k   = kb + i;
            int cin = k & 127;
            outp[i] = (_Float16)(kern[k * 128 + n] * he_std * s0_lds[cin] * dn);
        }
    }
}

// A-fragment gather (contiguous k order, same as wfrag B): elems 0..15 -> K=g*16+i
__device__ __forceinline__ v16h afrag_load(const _Float16* row, int cin0) {
    v8h lo = *(const v8h*)(row + cin0);
    v8h hi = *(const v8h*)(row + cin0 + 8);
    return __builtin_shufflevector(lo, hi,
        0, 1, 2, 3, 4, 5, 6, 7, 8, 9, 10, 11, 12, 13, 14, 15);
}

// ---------------- Main: implicit-GEMM conv via v_wmma_f32_16x16x32_f16 ----------------
// Block = 256 threads (8 waves). Block tile: 256 pixels (one image row) x 128 cout.
// Each wave owns 32 M-rows (two 16-row tiles) so every B fragment feeds two WMMAs.
// LDS: padded halo patch [3][258][136] f16 + double-buffered per-tap weights (2 x 32 KB).
__global__ __launch_bounds__(256, 1) void modconv_main(
    const float* __restrict__ x, const _Float16* __restrict__ wfrag,
    const float* __restrict__ noise, const float* __restrict__ noise_strength,
    const float* __restrict__ bias, float* __restrict__ out)
{
    extern __shared__ __align__(32) _Float16 smem[];
    _Float16* patch = smem;                      // PATCH_HALVES
    _Float16* wbuf  = smem + PATCH_HALVES;       // 2 * WTAP_HALVES

    const int tid = threadIdx.x;
    const int bid = blockIdx.x;                  // 2048 blocks: b*256 + h
    const int b   = bid >> 8;
    const int h   = bid & 255;

    // ---- kick off tap-0 weight staging (async path overlaps with patch conversion) ----
#if defined(ASYNC_WSTAGE)
    wstage_async(wfrag, wbuf, tid);
#endif

    // ---- stage halo patch into LDS (f32 -> f16, packed v4h stores) ----
    for (int idx = tid; idx < 3 * PCOLS * 32; idx += 256) {
        int dh = idx / (PCOLS * 32);
        int r2 = idx - dh * (PCOLS * 32);
        int wl = r2 >> 5;
        int c4 = r2 & 31;
        int gh = h - 1 + dh;
        int gw = wl - 1;
        float4 v = make_float4(0.f, 0.f, 0.f, 0.f);
        if ((unsigned)gh < 256u && (unsigned)gw < 256u)
            v = *(const float4*)&x[(((size_t)b * 256 + gh) * 256 + gw) * 128 + c4 * 4];
        v4h hv = { (_Float16)v.x, (_Float16)v.y, (_Float16)v.z, (_Float16)v.w };
        *(v4h*)&patch[(dh * PCOLS + wl) * ROWSTRIDE + c4 * 4] = hv;
    }

#if defined(ASYNC_WSTAGE)
    wstage_wait();
#else
    {   // register relay for tap 0
        const uint4* src = (const uint4*)wfrag;
        uint4* dst = (uint4*)wbuf;
        #pragma unroll
        for (int j = 0; j < 8; ++j)
            dst[tid + j * 256] = src[tid + j * 256];
    }
#endif
    __syncthreads();

    const int lane = tid & 31;
    const int wv   = tid >> 5;        // wave id 0..7 -> M rows [wv*32, wv*32+32)
    const int g    = lane >> 4;
    const int mrow = lane & 15;
    const int p0   = wv * 32 + mrow;       // first 16-row tile
    const int p1   = wv * 32 + 16 + mrow;  // second 16-row tile

    v8f acc[16] = {};

    for (int kk = 0; kk < 9; ++kk) {
        const int buf = kk & 1;
        const v16h* wv16 = (const v16h*)(wbuf + buf * WTAP_HALVES);

        // prefetch next tap's weights straight into the alternate LDS buffer
#if defined(ASYNC_WSTAGE)
        if (kk < 8)
            wstage_async(wfrag + (size_t)(kk + 1) * WTAP_HALVES,
                         wbuf + (1 - buf) * WTAP_HALVES, tid);
#else
        uint4 pre[8];
        if (kk < 8) {
            const uint4* src = (const uint4*)(wfrag + (size_t)(kk + 1) * WTAP_HALVES);
            #pragma unroll
            for (int j = 0; j < 8; ++j) pre[j] = src[tid + j * 256];
        }
#endif

        const int dh = kk / 3;
        const int dw = kk - 3 * dh;
        const _Float16* row0 = &patch[(dh * PCOLS + p0 + dw) * ROWSTRIDE];
        const _Float16* row1 = &patch[(dh * PCOLS + p1 + dw) * ROWSTRIDE];

        // software-pipelined: A fragments rotate across ks, B fragments rotate across nt
        v16h a0 = afrag_load(row0, g * 16);
        v16h a1 = afrag_load(row1, g * 16);
        #pragma unroll
        for (int ks = 0; ks < 4; ++ks) {
            v16h n0, n1;
            if (ks < 3) {
                const int cin0 = (ks + 1) * 32 + g * 16;
                n0 = afrag_load(row0, cin0);
                n1 = afrag_load(row1, cin0);
            }
            const v16h* wp = wv16 + ks * 256 + lane;
            v16h bcur = wp[0];
            #pragma unroll
            for (int nt = 0; nt < 8; ++nt) {
                v16h bnext;
                if (nt < 7) bnext = wp[(nt + 1) * 32];
                acc[nt] = __builtin_amdgcn_wmma_f32_16x16x32_f16(
                    false, a0, false, bcur, (short)0, acc[nt], false, false);
                acc[8 + nt] = __builtin_amdgcn_wmma_f32_16x16x32_f16(
                    false, a1, false, bcur, (short)0, acc[8 + nt], false, false);
                if (nt < 7) bcur = bnext;
            }
            if (ks < 3) { a0 = n0; a1 = n1; }
        }

#if defined(ASYNC_WSTAGE)
        wstage_wait();
#else
        if (kk < 8) {
            uint4* dst = (uint4*)(wbuf + (1 - buf) * WTAP_HALVES);
            #pragma unroll
            for (int j = 0; j < 8; ++j) dst[tid + j * 256] = pre[j];
        }
#endif
        __syncthreads();
    }

    // ---- epilogue: noise, bias, lrelu(0.2)*sqrt(2) ----
    const float ns       = noise_strength[0] * 0.5f;
    const float SQRT2    = 1.4142135623730951f;
    const size_t pixbase = (size_t)bid * 256;
    #pragma unroll
    for (int t = 0; t < 2; ++t) {
        #pragma unroll
        for (int nt = 0; nt < 8; ++nt) {
            const int cout = nt * 16 + mrow;             // C/D layout: N = lane&15
            const float bval = bias[cout];
            #pragma unroll
            for (int r = 0; r < 8; ++r) {
                const int pp  = wv * 32 + t * 16 + g * 8 + r;  // M = r + 8*(lane>=16)
                const size_t pix = pixbase + pp;
                float y = acc[t * 8 + nt][r] + noise[pix] * ns + bval;
                y = (y >= 0.f ? y : 0.2f * y) * SQRT2;
                out[pix * 128 + cout] = y;
            }
        }
    }
}

extern "C" void kernel_launch(void* const* d_in, const int* in_sizes, int n_in,
                              void* d_out, int out_size, void* d_ws, size_t ws_size,
                              hipStream_t stream) {
    const float* x      = (const float*)d_in[0];
    const float* style  = (const float*)d_in[1];
    const float* kern   = (const float*)d_in[2];
    const float* w_mod  = (const float*)d_in[3];
    const float* b_mod  = (const float*)d_in[4];
    const float* nstr   = (const float*)d_in[5];
    const float* bias   = (const float*)d_in[6];
    const float* noise  = (const float*)d_in[7];
    float* out          = (float*)d_out;
    _Float16* wfrag     = (_Float16*)d_ws;          // 1152*128 halves = 294912 B

    modconv_prep<<<1, 256, 0, stream>>>(style, kern, w_mod, b_mod, wfrag);

    const size_t shmem = (PATCH_HALVES + 2 * WTAP_HALVES) * sizeof(_Float16); // 276064 B
    (void)hipFuncSetAttribute((const void*)modconv_main,
                              hipFuncAttributeMaxDynamicSharedMemorySize, (int)shmem);
    const int nblocks = 8 * 256;                     // B * H = 2048 (one image row each)
    modconv_main<<<nblocks, 256, shmem, stream>>>(x, wfrag, noise, nstr, bias, out);
}
